// Attention_65034394796187
// MI455X (gfx1250) — compile-verified, matrix operand order checked by reading, not run on hardware
//
#include <hip/hip_runtime.h>
#include <hip/hip_bf16.h>
#include <math.h>

typedef __attribute__((ext_vector_type(16))) _Float16 v16h;
typedef __attribute__((ext_vector_type(8)))  float    v8f;

#define BS   2
#define SEQ  2048
#define DIM  2048
#define NH   16
#define NKV  8
#define HD   128
#define KVD  (NKV * HD)     // 1024
#define MTOT (BS * SEQ)     // 4096

#if defined(__AMDGCN__) && __has_builtin(__builtin_amdgcn_tensor_load_to_lds)
#define HAVE_TDM 1
#else
#define HAVE_TDM 0
#endif

#if HAVE_TDM
typedef unsigned int __attribute__((ext_vector_type(4))) tdm4u;
typedef int          __attribute__((ext_vector_type(4))) tdm4i;
typedef int          __attribute__((ext_vector_type(8))) tdm8i;

// Issue a TDM 2D tile load: global (rows x cols of f16, row stride `stride`
// elements) -> LDS at byte offset lds_off, rows stored contiguously.
// D# packing per CDNA5 ISA ch.8: group0 = {flags, lds_addr, global_addr, type},
// group1 = {data_size, tensor dims (== tile dims, no OOB), tile dims, stride}.
__device__ __forceinline__ void tdm_load_2d(unsigned lds_off, const void* gptr,
                                            unsigned rows, unsigned cols,
                                            unsigned stride_elems) {
  unsigned long long ga = (unsigned long long)gptr;
  tdm4u g0;
  g0[0] = 1u;                                         // count=1, user descriptor
  g0[1] = lds_off;                                    // lds_addr (bytes)
  g0[2] = (unsigned)(ga & 0xFFFFFFFFull);             // global_addr[31:0]
  g0[3] = (unsigned)((ga >> 32) & 0x1FFFFFFull)       // global_addr[56:32]
          | (2u << 30);                               // type = 2 ("image")
  tdm8i g1;
  g1[0] = (int)(1u << 16);                            // data_size=1 (2 bytes)
  g1[1] = (int)((cols & 0xFFFFu) << 16);              // tensor_dim0[15:0]
  g1[2] = (int)(((cols >> 16) & 0xFFFFu) | ((rows & 0xFFFFu) << 16));
  g1[3] = (int)(((rows >> 16) & 0xFFFFu) | ((cols & 0xFFFFu) << 16)); // tile_dim0
  g1[4] = (int)(rows & 0xFFFFu);                      // tile_dim1, tile_dim2=0
  g1[5] = (int)stride_elems;                          // tensor_dim0_stride[31:0]
  g1[6] = 0;
  g1[7] = 0;
  tdm4i z4 = {0, 0, 0, 0};
#if defined(__clang_major__) && __clang_major__ >= 23
  tdm8i z8 = {0, 0, 0, 0, 0, 0, 0, 0};
  __builtin_amdgcn_tensor_load_to_lds(g0, g1, z4, z4, z8, 0);
#else
  __builtin_amdgcn_tensor_load_to_lds(g0, g1, z4, z4, 0);
#endif
}
#endif  // HAVE_TDM

// ---------------------------------------------------------------------------
// Fragment loaders following the CDNA5 ISA 16-bit WMMA VGPR layouts (wave32).
// A-matrix 16x32 f16: lanes 0-15 -> M=lane, VGPR i<4 holds K = 2i,2i+1;
// lanes 16-31 -> same M, K offset by 8; VGPRs 4-7 repeat with K += 16.
// ---------------------------------------------------------------------------
__device__ __forceinline__ v16h load_a_frag(const _Float16* __restrict__ base,
                                            int ld, int m0, int k0, int lane) {
  const int m  = m0 + (lane & 15);
  const int hf = lane >> 4;
  const _Float16* rp = base + (size_t)m * ld + k0;
  union { v16h v; unsigned int u[8]; } r;
#pragma unroll
  for (int i = 0; i < 4; ++i)
    r.u[i] = *(const unsigned int*)(rp + hf * 8 + 2 * i);
#pragma unroll
  for (int i = 0; i < 4; ++i)
    r.u[4 + i] = *(const unsigned int*)(rp + 16 + hf * 8 + 2 * i);
  return r.v;
}

// ---------------------------------------------------------------------------
// Prep kernels: f32->f16 convert; transposing variants so every hot-loop tile
// becomes a plain 2D strided copy the TDM can move (no SIMD transposes).
// ---------------------------------------------------------------------------
__global__ void cvt_f32_to_f16(const float* __restrict__ src,
                               _Float16* __restrict__ dst, int n) {
  int i = blockIdx.x * blockDim.x + threadIdx.x;
  if (i < n) dst[i] = (_Float16)src[i];
}

// W (DIM x Nout, f32) -> Wt (Nout x DIM, f16); writes coalesced along K.
__global__ void cvt_transpose_f32_to_f16(const float* __restrict__ W,
                                         _Float16* __restrict__ Wt, int Nout) {
  const int idx = blockIdx.x * blockDim.x + threadIdx.x;
  const int k = idx & (DIM - 1);
  const int n = idx >> 11;                  // DIM == 2048
  if (n < Nout) Wt[(size_t)n * DIM + k] = (_Float16)W[(size_t)k * Nout + n];
}

// Vh (MTOT x KVD, f16) -> Vt [(b*KVD + c)*SEQ + s]; writes coalesced along s.
__global__ void transpose_v_f16(const _Float16* __restrict__ Vh,
                                _Float16* __restrict__ Vt) {
  const int idx = blockIdx.x * blockDim.x + threadIdx.x;
  if (idx >= MTOT * KVD) return;
  const int s = idx & (SEQ - 1);
  const int c = (idx >> 11) & (KVD - 1);
  const int b = idx >> 21;
  Vt[idx] = Vh[((size_t)(b * SEQ + s)) * KVD + c];
}

// ---------------------------------------------------------------------------
// GEMM: Y[M x Nout] = A[M x DIM] * Wt^T + bias, f16 in/out, f32 acc.
// Wt is pre-transposed (Nout x DIM) so both tiles are straight 2D copies.
// Block: 256 threads = 8 waves; block tile 128(M) x 64(N); wave tile 32x32;
// K-step 64, double-buffered: wave 0 issues next tile pair via the TDM while
// all waves run WMMA on the current pair (TENSORcnt completes in-order, so
// s_wait_tensorcnt(2) guarantees the current pair has landed).
// ---------------------------------------------------------------------------
__global__ __launch_bounds__(256) void gemm_bias_f16(
    const _Float16* __restrict__ A,
    const _Float16* __restrict__ Wt,    // Nout x DIM
    const float*    __restrict__ bias,
    _Float16*       __restrict__ Y,
    int Nout) {
  __shared__ _Float16 lA[2][128][64];   // M x K tiles
  __shared__ _Float16 lBt[2][64][64];   // N x K tiles

  const int tid   = threadIdx.x;
  const int lane  = tid & 31;
  const int w     = tid >> 5;
  const int waveM = w & 3;
  const int waveN = w >> 2;
  const int m0    = blockIdx.y * 128;
  const int n0    = blockIdx.x * 64;

  v8f acc[2][2] = {};
  int cur = 0;

#if HAVE_TDM
  if (w == 0) {
    tdm_load_2d((unsigned)(unsigned long long)&lA[0][0][0],
                &A[(size_t)m0 * DIM], 128, 64, DIM);
    tdm_load_2d((unsigned)(unsigned long long)&lBt[0][0][0],
                &Wt[(size_t)n0 * DIM], 64, 64, DIM);
  }
#endif

  for (int k0 = 0; k0 < DIM; k0 += 64) {
    const bool hasNext = (k0 + 64 < DIM);
#if HAVE_TDM
    if (w == 0) {
      if (hasNext) {
        tdm_load_2d((unsigned)(unsigned long long)&lA[cur ^ 1][0][0],
                    &A[(size_t)m0 * DIM + k0 + 64], 128, 64, DIM);
        tdm_load_2d((unsigned)(unsigned long long)&lBt[cur ^ 1][0][0],
                    &Wt[(size_t)n0 * DIM + k0 + 64], 64, 64, DIM);
        __builtin_amdgcn_s_wait_tensorcnt(2);
      } else {
        __builtin_amdgcn_s_wait_tensorcnt(0);
      }
    }
#else
#pragma unroll
    for (int it = 0; it < 4; ++it) {
      const int li = tid + it * 256;           // 0..1023 quads
      const int r  = li >> 3;
      const int c  = (li & 7) * 8;
      *(uint4*)&lA[cur][r][c] = *(const uint4*)&A[(size_t)(m0 + r) * DIM + k0 + c];
    }
#pragma unroll
    for (int it = 0; it < 2; ++it) {
      const int li = tid + it * 256;           // 0..511 quads
      const int nn = li >> 3;
      const int kq = (li & 7) * 8;
      *(uint4*)&lBt[cur][nn][kq] = *(const uint4*)&Wt[(size_t)(n0 + nn) * DIM + k0 + kq];
    }
#endif
    if (k0 + 128 < DIM)   // next tile is in flight; warm L2 for the one after
      __builtin_prefetch(&A[(size_t)(m0 + (tid >> 3)) * DIM + k0 + 128], 0, 3);
    __syncthreads();

#pragma unroll
    for (int kk = 0; kk < 64; kk += 32) {
      v16h aF[2], bF[2];
#pragma unroll
      for (int i = 0; i < 2; ++i)
        aF[i] = load_a_frag(&lA[cur][0][0], 64, waveM * 32 + i * 16, kk, lane);
#pragma unroll
      for (int j = 0; j < 2; ++j)
        bF[j] = load_a_frag(&lBt[cur][0][0], 64, waveN * 32 + j * 16, kk, lane);
#pragma unroll
      for (int i = 0; i < 2; ++i)
#pragma unroll
        for (int j = 0; j < 2; ++j)
          acc[i][j] = __builtin_amdgcn_wmma_f32_16x16x32_f16(
              false, aF[i], false, bF[j], (short)0, acc[i][j], false, false);
    }
    __syncthreads();
    cur ^= 1;
  }

  // Epilogue: C layout -> element e holds M = e + 8*(lane>>4), N = lane&15.
  const int hf = lane >> 4;
  const int ln = lane & 15;
#pragma unroll
  for (int i = 0; i < 2; ++i)
#pragma unroll
    for (int j = 0; j < 2; ++j)
#pragma unroll
      for (int e = 0; e < 8; ++e) {
        const int row = m0 + waveM * 32 + i * 16 + e + 8 * hf;
        const int col = n0 + waveN * 32 + j * 16 + ln;
        const float yv = acc[i][j][e] + bias[col];
        Y[(size_t)row * Nout + col] = (_Float16)yv;
      }
}

// ---------------------------------------------------------------------------
// RoPE on interleaved pairs, f32 math, f16 storage, in place.
// ---------------------------------------------------------------------------
__global__ void rope_f16(_Float16* __restrict__ Y, int nheads) {
  const int total = MTOT * nheads * (HD / 2);
  const int idx = blockIdx.x * blockDim.x + threadIdx.x;
  if (idx >= total) return;
  const int i   = idx & 63;                 // HD/2 == 64
  const int h   = (idx >> 6) % nheads;
  const int row = idx / (nheads * 64);
  const int pos = row & (SEQ - 1);
  _Float16* p = Y + (size_t)row * nheads * HD + h * HD + 2 * i;
  const float inv = __powf(10000.0f, -(2.0f * (float)i) / 128.0f);
  const float f = (float)pos * inv;
  const float c = cosf(f), s = sinf(f);
  const float xr = (float)p[0], xi = (float)p[1];
  p[0] = (_Float16)(xr * c - xi * s);
  p[1] = (_Float16)(xr * s + xi * c);
}

// ---------------------------------------------------------------------------
// Flash attention, causal, GQA. Block = 256 threads (8 waves) covers 128 query
// rows of one (batch, head). Per 32-key step the K tile (key-major) and the
// pre-transposed V tile (d-major) are both moved by the TDM, double-buffered
// against the WMMA work. Online softmax in f32 registers; P staged through
// per-wave LDS to reshape into the A-fragment layout.
// ---------------------------------------------------------------------------
__global__ __launch_bounds__(256) void flash_attn(
    const _Float16* __restrict__ Qh,   // MTOT x (NH*HD)
    const _Float16* __restrict__ Kh,   // MTOT x KVD
    const _Float16* __restrict__ Vt,   // (BS*KVD) x SEQ  (d-major V)
    float*          __restrict__ out)  // (bs, seq, NH, HD)
{
  __shared__ _Float16 lK[2][32][128];   // key-major K tiles
  __shared__ _Float16 lVt[2][128][32];  // d-major V tiles
  __shared__ _Float16 pbuf[8][16][32];  // per-wave probs staging

  const int tid   = threadIdx.x;
  const int lane  = tid & 31;
  const int w     = tid >> 5;
  const int hf    = lane >> 4;
  const int ln    = lane & 15;
  const int qTile = blockIdx.x * 128;
  const int qBase = qTile + w * 16;
  const int h     = blockIdx.y;
  const int b     = blockIdx.z;
  const int kvh   = h / (NH / NKV);

  const _Float16* Qb  = Qh + ((size_t)b * SEQ) * (NH * HD) + h * HD;
  const _Float16* Kb  = Kh + ((size_t)b * SEQ) * KVD + kvh * HD;
  const _Float16* VtB = Vt + ((size_t)b * KVD + kvh * HD) * SEQ;  // row d, col s

  // Q fragments for head_dim 128 = 4 chunks of K=32, kept in registers.
  v16h qF[4];
#pragma unroll
  for (int t = 0; t < 4; ++t)
    qF[t] = load_a_frag(Qb, NH * HD, qBase, t * 32, lane);

  v8f o[8] = {};
  float m[8], l[8];
#pragma unroll
  for (int e = 0; e < 8; ++e) { m[e] = -1e30f; l[e] = 0.0f; }

  const float scale = 1.0f / sqrtf((float)DIM);   // reference scales by sqrt(dim)
  const int nkb = (qTile + 127) / 32 + 1;         // causal limit for the block
  int cur = 0;

#if HAVE_TDM
  if (w == 0) {
    tdm_load_2d((unsigned)(unsigned long long)&lK[0][0][0], Kb, 32, 128, KVD);
    tdm_load_2d((unsigned)(unsigned long long)&lVt[0][0][0], VtB, 128, 32, SEQ);
  }
#endif

  for (int kb = 0; kb < nkb; ++kb) {
    const int key0 = kb * 32;
    const bool hasNext = (kb + 1 < nkb);
#if HAVE_TDM
    if (w == 0) {
      if (hasNext) {
        tdm_load_2d((unsigned)(unsigned long long)&lK[cur ^ 1][0][0],
                    &Kb[(size_t)(key0 + 32) * KVD], 32, 128, KVD);
        tdm_load_2d((unsigned)(unsigned long long)&lVt[cur ^ 1][0][0],
                    &VtB[key0 + 32], 128, 32, SEQ);
        __builtin_amdgcn_s_wait_tensorcnt(2);
      } else {
        __builtin_amdgcn_s_wait_tensorcnt(0);
      }
    }
#else
#pragma unroll
    for (int it = 0; it < 2; ++it) {
      const int li  = tid + it * 256;           // 0..511 quads
      const int key = li >> 4;
      const int d0  = (li & 15) * 8;
      *(uint4*)&lK[cur][key][d0] = *(const uint4*)&Kb[(size_t)(key0 + key) * KVD + d0];
    }
#pragma unroll
    for (int it = 0; it < 2; ++it) {
      const int li = tid + it * 256;            // 0..511 quads
      const int d  = li >> 2;
      const int kq = (li & 3) * 8;
      *(uint4*)&lVt[cur][d][kq] = *(const uint4*)&VtB[(size_t)d * SEQ + key0 + kq];
    }
#endif
    __syncthreads();

    if (key0 <= qBase + 15) {
      // S = Q * K^T for 32 keys -> two 16x16 f32 tiles (B frags from lK).
      v8f s[2] = {};
#pragma unroll
      for (int sub = 0; sub < 2; ++sub)
#pragma unroll
        for (int t = 0; t < 4; ++t) {
          v16h bF = load_a_frag(&lK[cur][0][0], 128, sub * 16, t * 32, lane);
          s[sub] = __builtin_amdgcn_wmma_f32_16x16x32_f16(
              false, qF[t], false, bF, (short)0, s[sub], false, false);
        }

      // Online softmax: rows of a C tile live within one 16-lane half-wave.
#pragma unroll
      for (int e = 0; e < 8; ++e) {
        const int q = qBase + e + 8 * hf;
        float s0 = s[0][e] * scale;
        float s1 = s[1][e] * scale;
        if (key0 + ln > q)      s0 = -1e30f;
        if (key0 + 16 + ln > q) s1 = -1e30f;

        float v = fmaxf(s0, s1);
#pragma unroll
        for (int off = 1; off < 16; off <<= 1)
          v = fmaxf(v, __shfl_xor(v, off, 32));
        const float mn    = fmaxf(m[e], v);
        const float alpha = __expf(m[e] - mn);
        const float p0 = __expf(s0 - mn);
        const float p1 = __expf(s1 - mn);
        float rs = p0 + p1;
#pragma unroll
        for (int off = 1; off < 16; off <<= 1)
          rs += __shfl_xor(rs, off, 32);
        l[e] = l[e] * alpha + rs;
        m[e] = mn;
#pragma unroll
        for (int t = 0; t < 8; ++t) o[t][e] = o[t][e] * alpha;

        pbuf[w][e + 8 * hf][ln]      = (_Float16)p0;
        pbuf[w][e + 8 * hf][16 + ln] = (_Float16)p1;
      }
      // Intra-wave LDS RAW: lockstep wave32, just drain the DS counter.
      asm volatile("s_wait_dscnt 0" ::: "memory");

      // O += P * V  (P: 16x32 A-frag from pbuf; V frags contiguous in lVt).
      v16h pF = load_a_frag(&pbuf[w][0][0], 32, 0, 0, lane);
#pragma unroll
      for (int t = 0; t < 8; ++t) {
        v16h vF = load_a_frag(&lVt[cur][0][0], 32, t * 16, 0, lane);
        o[t] = __builtin_amdgcn_wmma_f32_16x16x32_f16(
            false, pF, false, vF, (short)0, o[t], false, false);
      }
    }
    __syncthreads();
    cur ^= 1;
  }

#pragma unroll
  for (int t = 0; t < 8; ++t)
#pragma unroll
    for (int e = 0; e < 8; ++e) {
      const int q = qBase + e + 8 * hf;
      const int d = t * 16 + ln;
      out[(((size_t)(b * SEQ) + q) * NH + h) * HD + d] = o[t][e] / l[e];
    }
}

// ---------------------------------------------------------------------------
// Launch
// ---------------------------------------------------------------------------
extern "C" void kernel_launch(void* const* d_in, const int* in_sizes, int n_in,
                              void* d_out, int out_size, void* d_ws, size_t ws_size,
                              hipStream_t stream) {
  (void)in_sizes; (void)n_in; (void)out_size; (void)ws_size;

  const float* hidden = (const float*)d_in[0];
  const float* wq     = (const float*)d_in[1];
  const float* bq     = (const float*)d_in[2];
  const float* wk     = (const float*)d_in[3];
  const float* bk     = (const float*)d_in[4];
  const float* wv     = (const float*)d_in[5];
  const float* bv     = (const float*)d_in[6];
  float* out = (float*)d_out;

  // Workspace layout (72 MB total, f16 buffers).
  char* ws = (char*)d_ws;
  _Float16* Xh  = (_Float16*)ws; ws += (size_t)MTOT * DIM * 2;   // 16 MB
  _Float16* Wqt = (_Float16*)ws; ws += (size_t)DIM * DIM * 2;    //  8 MB (transposed)
  _Float16* Wkt = (_Float16*)ws; ws += (size_t)DIM * KVD * 2;    //  4 MB (transposed)
  _Float16* Wvt = (_Float16*)ws; ws += (size_t)DIM * KVD * 2;    //  4 MB (transposed)
  _Float16* Qh  = (_Float16*)ws; ws += (size_t)MTOT * DIM * 2;   // 16 MB
  _Float16* Kh  = (_Float16*)ws; ws += (size_t)MTOT * KVD * 2;   //  8 MB
  _Float16* Vh  = (_Float16*)ws; ws += (size_t)MTOT * KVD * 2;   //  8 MB
  _Float16* Vt  = (_Float16*)ws; ws += (size_t)MTOT * KVD * 2;   //  8 MB (d-major)

  const int nX  = MTOT * DIM;
  const int nWq = DIM * DIM;
  const int nWk = DIM * KVD;
  cvt_f32_to_f16<<<(nX + 255) / 256, 256, 0, stream>>>(hidden, Xh, nX);
  cvt_transpose_f32_to_f16<<<(nWq + 255) / 256, 256, 0, stream>>>(wq, Wqt, DIM);
  cvt_transpose_f32_to_f16<<<(nWk + 255) / 256, 256, 0, stream>>>(wk, Wkt, KVD);
  cvt_transpose_f32_to_f16<<<(nWk + 255) / 256, 256, 0, stream>>>(wv, Wvt, KVD);

  dim3 gq(DIM / 64, MTOT / 128);
  dim3 gkv(KVD / 64, MTOT / 128);
  gemm_bias_f16<<<gq,  256, 0, stream>>>(Xh, Wqt, bq, Qh, DIM);
  gemm_bias_f16<<<gkv, 256, 0, stream>>>(Xh, Wkt, bk, Kh, KVD);
  gemm_bias_f16<<<gkv, 256, 0, stream>>>(Xh, Wvt, bv, Vh, KVD);

  const int nRq = MTOT * NH  * (HD / 2);
  const int nRk = MTOT * NKV * (HD / 2);
  rope_f16<<<(nRq + 255) / 256, 256, 0, stream>>>(Qh, NH);
  rope_f16<<<(nRk + 255) / 256, 256, 0, stream>>>(Kh, NKV);

  transpose_v_f16<<<(MTOT * KVD + 255) / 256, 256, 0, stream>>>(Vh, Vt);

  dim3 ga(SEQ / 128, NH, BS);
  flash_attn<<<ga, 256, 0, stream>>>(Qh, Kh, Vt, out);
}
